// Prop_padding_8203387535683
// MI455X (gfx1250) — matrix operand order repaired and checked by reading, not compiled, not run on hardware
//
#include <hip/hip_runtime.h>
#include <cmath>

typedef __attribute__((ext_vector_type(16))) __bf16 v16bf;
typedef __attribute__((ext_vector_type(8)))  __bf16 v8bf;
typedef __attribute__((ext_vector_type(2)))  __bf16 v2bf;
typedef __attribute__((ext_vector_type(8)))  float  v8f;

union F16 { v16bf v; v8bf h[2]; };

#define TL 64                 // output positions per workgroup
#define KTOTP_MAX 224         // max K (Cin*KW) padded to 32 (sr conv: 40*5=200->224)
#define PITCH_MAX (KTOTP_MAX + 8)   // +8 halfs: keeps 16B alignment, breaks bank stride
#define COUTP_MAX 64          // max Cout padded to 16

__device__ __forceinline__ v2bf pack2(float a, float b) {
  v2bf t; t[0] = (__bf16)a; t[1] = (__bf16)b; return t;   // v_cvt_pk_bf16_f32
}

// ---------------------------------------------------------------------------
// 1-D conv as GEMM on the WMMA pipe, LDS-staged.
// WG = 256 threads (8 waves): one batch row x TL output positions.
//  stage1: im2col input tile + weight panel -> LDS bf16 (pair-packed stores)
//  stage2: fragments via 2x ds_load_b128 each (contiguous 8-half runs) + WMMA.
// ---------------------------------------------------------------------------
template<int KW>
__global__ __launch_bounds__(256) void conv1d_wmma_bf16(
    const float* __restrict__ in, int cin_total, int cin_off, int L_in,
    const float* __restrict__ W, const float* __restrict__ bias,
    float* __restrict__ out, int cout_total, int cout_off, int L_out,
    const float* __restrict__ res, int res_ctotal, int res_coff,
    int Cin, int Cout, int stride, int pad, int relu)
{
  __shared__ __align__(16) __bf16 lds_x[TL * PITCH_MAX];        // [l_local][kk]
  __shared__ __align__(16) __bf16 lds_w[COUTP_MAX * PITCH_MAX]; // [co][kk]

  const int tid   = threadIdx.x;
  const int lane  = tid & 31;
  const int wave  = tid >> 5;
  const int Ktot  = Cin * KW;
  const int KtotP = (Ktot + 31) & ~31;
  const int pitch = KtotP + 8;
  const int CoutP = (Cout + 15) & ~15;
  const int nCT   = CoutP >> 4;

  const int nLB = L_out / TL;
  const int b   = blockIdx.x / nLB;
  const int lb  = blockIdx.x - b * nLB;
  const int l0g = lb * TL;

  const float* inB = in + ((long long)b * cin_total + cin_off) * L_in;

  // ---- cooperative fill: weight panel (no divides, pair-packed) ----
  for (int co = wave; co < CoutP; co += 8) {
    const int coc = co < Cout ? co : Cout - 1;
    const float* wr = W + (long long)coc * Ktot;
    const bool cok = co < Cout;
    for (int kk = lane * 2; kk < KtotP; kk += 64) {
      int k0 = kk     < Ktot ? kk     : Ktot - 1;
      int k1 = kk + 1 < Ktot ? kk + 1 : Ktot - 1;
      float v0 = (cok && kk     < Ktot) ? wr[k0] : 0.f;
      float v1 = (cok && kk + 1 < Ktot) ? wr[k1] : 0.f;
      *reinterpret_cast<v2bf*>(&lds_w[co * pitch + kk]) = pack2(v0, v1);
    }
  }
  // ---- cooperative fill: im2col tile (no divides, pair-packed) ----
  for (int ll = wave; ll < TL; ll += 8) {
    const int lbase = (l0g + ll) * stride - pad;
    for (int kk = lane * 2; kk < KtotP; kk += 64) {
      float v[2];
      #pragma unroll
      for (int e = 0; e < 2; ++e) {
        int k = kk + e;
        int ci = k / KW;                    // KW constant -> strength-reduced
        int kw = k - ci * KW;
        int li = lbase + kw;
        bool ok = (k < Ktot) & (li >= 0) & (li < L_in);
        int lic = li < 0 ? 0 : (li >= L_in ? L_in - 1 : li);
        int cic = ci < Cin ? ci : Cin - 1;
        float t = inB[(long long)cic * L_in + lic];
        v[e] = ok ? t : 0.f;
      }
      *reinterpret_cast<v2bf*>(&lds_x[ll * pitch + kk]) = pack2(v[0], v[1]);
    }
  }
  __syncthreads();

  const int h  = (lane >> 4) & 1;
  const int mn = lane & 15;
  const int nKC = KtotP >> 5;
  const int nTiles = nCT * (TL / 16);

  for (int t = wave; t < nTiles; t += 8) {
    const int ct  = t / (TL / 16);
    const int ltl = t - ct * (TL / 16);
    const __bf16* wrow = &lds_w[(ct * 16 + mn) * pitch];
    const __bf16* xrow = &lds_x[(ltl * 16 + mn) * pitch];
    v8f acc = {0.f,0.f,0.f,0.f,0.f,0.f,0.f,0.f};
    for (int kc = 0; kc < nKC; ++kc) {
      const int kbase = kc << 5;
      F16 a, bm;
      // A 16x32: elems[0..7]=K kbase+8h+0..7, elems[8..15]=K kbase+16+8h+0..7
      const __bf16* wp = wrow + kbase + 8 * h;
      a.h[0] = *reinterpret_cast<const v8bf*>(wp);
      a.h[1] = *reinterpret_cast<const v8bf*>(wp + 16);
      // B 32x16: elems[0..7]=K kbase+8h+0..7, elems[8..15]=K kbase+16+8h+0..7
      const __bf16* xp = xrow + kbase + 8 * h;
      bm.h[0] = *reinterpret_cast<const v8bf*>(xp);
      bm.h[1] = *reinterpret_cast<const v8bf*>(xp + 16);
      acc = __builtin_amdgcn_wmma_f32_16x16x32_bf16(false, a.v, false, bm.v,
                                                    (short)0, acc, false, false);
    }
    // C/D layout: VGPR r -> M = r + 8*h, N = lane&15
    #pragma unroll
    for (int r = 0; r < 8; ++r) {
      int c = ct*16 + r + 8*h;
      if (c < Cout) {
        int l = l0g + ltl*16 + mn;
        float v = acc[r] + bias[c];
        if (res) v += res[((long long)b * res_ctotal + res_coff + c) * L_out + l];
        if (relu) v = fmaxf(v, 0.f);
        out[((long long)b * cout_total + cout_off + c) * L_out + l] = v;
      }
    }
  }
}

// ---------------------------------------------------------------------------
// FC GEMM + tanh:  out[b,j] = tanh( sum_i act[b,i]*W[i,j] + bias[j] )
// K % 32 == 0, B/N multiples of 16 -> fully guard-free inner loop.
// ---------------------------------------------------------------------------
__global__ __launch_bounds__(256) void fc_tanh_wmma_bf16(
    const float* __restrict__ act, const float* __restrict__ Wm,
    const float* __restrict__ bias, float* __restrict__ out,
    int Brows, int K, int N)
{
  const int lane = threadIdx.x & 31;
  const int wave = threadIdx.x >> 5;
  const int nBT = Brows >> 4, nNT = N >> 4;
  const long long tiles = (long long)nBT * nNT;
  long long tile = (long long)blockIdx.x * 8 + wave;
  if (tile >= tiles) return;
  const int nt = (int)(tile % nNT);
  const int bt = (int)(tile / nNT);
  const int h  = (lane >> 4) & 1;
  const int mn = lane & 15;
  const float* arow = act + (long long)(bt*16 + mn) * K;
  const float* wcol = Wm + nt*16 + mn;
  const int nKC = K >> 5;
  v8f acc = {0.f,0.f,0.f,0.f,0.f,0.f,0.f,0.f};
  for (int kc = 0; kc < nKC; ++kc) {
    const int kbase = kc << 5;
    v16bf afrag, bfrag;
    #pragma unroll
    for (int v = 0; v < 8; ++v) {
      const int kA = (v < 4) ? (8*h + 2*v) : (16 + 8*h + 2*(v-4));
      const int kB = (v < 4) ? (8*h + 2*v) : (16 + 8*h + 2*(v-4));
      float2 a2 = *reinterpret_cast<const float2*>(arow + kbase + kA);
      afrag[2*v]   = (__bf16)a2.x;
      afrag[2*v+1] = (__bf16)a2.y;
      bfrag[2*v]   = (__bf16)wcol[(long long)(kbase + kB    ) * N];
      bfrag[2*v+1] = (__bf16)wcol[(long long)(kbase + kB + 1) * N];
    }
    acc = __builtin_amdgcn_wmma_f32_16x16x32_bf16(false, afrag, false, bfrag,
                                                  (short)0, acc, false, false);
  }
  #pragma unroll
  for (int r = 0; r < 8; ++r) {
    int b = bt*16 + r + 8*h;
    int j = nt*16 + mn;
    out[(long long)b * N + j] = tanhf(acc[r] + bias[j]);
  }
}

// ---------------------------------------------------------------------------
// code_assign: per (b,l) softmax over M centroids; prob -> d_out slice,
// q -> channel slot of pre-zeroed concat tensor.
// ---------------------------------------------------------------------------
template<int M>
__global__ __launch_bounds__(256) void code_assign_kernel(
    const float* __restrict__ c, const float* __restrict__ means,
    float* __restrict__ prob, float* __restrict__ q,
    int Bn, int L, int q_ctotal, int q_coff, float scale)
{
  long long idx = (long long)blockIdx.x * blockDim.x + threadIdx.x;
  if (idx >= (long long)Bn * L) return;
  int b = (int)(idx / L), l = (int)(idx - (long long)b * L);
  float cv[15]; float cc = 0.f;
  #pragma unroll
  for (int d = 0; d < 15; ++d) {
    cv[d] = c[((long long)b*15 + d)*L + l];
    cc += cv[d]*cv[d];
  }
  float xs[M]; float xmax = -3.0e38f;
  for (int m = 0; m < M; ++m) {
    float dot = 0.f, msq = 0.f;
    #pragma unroll
    for (int d = 0; d < 15; ++d) { float mv = means[d*M+m]; dot += cv[d]*mv; msq += mv*mv; }
    xs[m] = -(cc + msq - 2.f*dot) * scale;
    xmax = fmaxf(xmax, xs[m]);
  }
  float s = 0.f;
  for (int m = 0; m < M; ++m) { xs[m] = expf(xs[m]-xmax); s += xs[m]; }
  float inv = 1.f / s;
  for (int m = 0; m < M; ++m) { xs[m] *= inv; prob[idx*M + m] = xs[m]; }
  #pragma unroll
  for (int d = 0; d < 15; ++d) {
    float qd = 0.f;
    for (int m = 0; m < M; ++m) qd += xs[m]*means[d*M+m];
    q[((long long)b*q_ctotal + q_coff + d)*L + l] = qd;
  }
}

__global__ void fill_zero(float* __restrict__ p, long long n) {
  long long i = (long long)blockIdx.x * blockDim.x + threadIdx.x;
  if (i < n) p[i] = 0.f;
}

// sub_pixel: in (B, 2*Ch, Lin) -> out channel slot (B, cout_total, 2*Lin)
__global__ void sub_pixel_kernel(const float* __restrict__ in, float* __restrict__ out,
                                 int Bn, int Ch, int Lin, int cout_total, int cout_off)
{
  long long idx = (long long)blockIdx.x * blockDim.x + threadIdx.x;
  long long total = (long long)Bn * Ch * Lin * 2;
  if (idx >= total) return;
  int t = (int)(idx & 1);
  long long r = idx >> 1;
  int l = (int)(r % Lin); r /= Lin;
  int c = (int)(r % Ch);
  int b = (int)(r / Ch);
  out[((long long)b*cout_total + cout_off + c)*(2*Lin) + 2*l + t] =
      in[((long long)b*(2*Ch) + 2*c + t)*Lin + l];
}

// ---------------------------------------------------------------------------
extern "C" void kernel_launch(void* const* d_in, const int* in_sizes, int n_in,
                              void* d_out, int out_size, void* d_ws, size_t ws_size,
                              hipStream_t stream)
{
  (void)in_sizes; (void)n_in; (void)out_size; (void)ws_size;
  enum {
    I_X=0, I_E0_W1, I_E0_B1, I_E0_W2, I_E0_B2, I_SR_W, I_SR_B,
    I_E1_W1, I_E1_B1, I_E1_W2, I_E1_B2,
    I_E2_W1, I_E2_B1, I_E2_W2, I_E2_B2,
    I_M0_W1, I_M0_B1, I_M0_W2, I_M0_B2,
    I_MC_W, I_MC_B,
    I_M1_W1, I_M1_B1, I_M1_W2, I_M1_B2,
    I_MEANS_S, I_MEANS_N, I_AI_W, I_AI_B,
    I_AB_W1, I_AB_B1, I_AB_W2, I_AB_B2,
    I_AO0_W1, I_AO0_B1, I_AO0_W2, I_AO0_B2,
    I_AO1_W1, I_AO1_B1, I_AO1_W2, I_AO1_B2,
    I_D0_W1, I_D0_B1, I_D0_W2, I_D0_B2,
    I_D1_W1, I_D1_B1, I_D1_W2, I_D1_B2,
    I_FC_W, I_FC_B
  };
  #define P(i) ((const float*)d_in[(i)])
  const int NB = 1024;
  float* outp = (float*)d_out;
  float* DEC_S  = outp + 0;
  float* DEC_N  = outp + 524288;      // 1024*512
  float* PROB_S = outp + 1048576;     // + (1024,256,32)
  float* PROB_N = outp + 9437184;

  // ---- workspace bump allocator (f32 elements), with aliasing ----
  float* ws = (float*)d_ws;
  size_t off = 0;
  auto alloc = [&](size_t n) { float* p = ws + off; off += n; return p; };
  const size_t SZ50_512 = (size_t)NB*50*512;
  const size_t SZ25_512 = (size_t)NB*25*512;
  float* CODE = alloc(SZ50_512);               // early: e0 tmp (B,40,512)
  float* TC1  = alloc(SZ50_512);               // early: e0 out (B,40,512)
  float* TC2  = alloc(SZ50_512);               // early: H1|H2 (2x B,40,256)
  float* R50a = alloc(SZ25_512);               // early: e1/e2 tmp; addin A50; dec D1
  float* R50b = alloc(SZ25_512);               // addin B50; dec D2
  float* R50c = alloc(SZ25_512);               // addin C50; dec D3
  float* A20  = alloc((size_t)NB*20*256);
  float* B20  = alloc((size_t)NB*20*256);
  float* A15  = alloc((size_t)NB*15*256);
  float* B15  = alloc((size_t)NB*15*256);
  float* C15  = alloc((size_t)NB*15*256);
  float* C30s = alloc((size_t)NB*30*256);
  float* C30n = alloc((size_t)NB*30*256);
  float* H1 = TC2;
  float* H2 = TC2 + (size_t)NB*40*256;

  auto conv = [&](int KW, const float* in, int cint, int cinoff, int Lin,
                  const float* W, const float* bias,
                  float* out, int coutt, int coutoff, int Lout,
                  const float* res, int rest, int resoff,
                  int Cin, int Cout, int stride, int pad, int relu) {
    int blocks = NB * (Lout / TL);
    if (KW == 3)
      conv1d_wmma_bf16<3><<<blocks,256,0,stream>>>(in,cint,cinoff,Lin,W,bias,
          out,coutt,coutoff,Lout,res,rest,resoff,Cin,Cout,stride,pad,relu);
    else
      conv1d_wmma_bf16<5><<<blocks,256,0,stream>>>(in,cint,cinoff,Lin,W,bias,
          out,coutt,coutoff,Lout,res,rest,resoff,Cin,Cout,stride,pad,relu);
  };
  auto fill0 = [&](float* p, size_t n) {
    fill_zero<<<(int)((n+255)/256),256,0,stream>>>(p, (long long)n);
  };

  const float* x = P(I_X);   // (1024, 512) == (B, 1, 512)

  // ---- encoder ----
  conv(3, x,1,0,512,    P(I_E0_W1),P(I_E0_B1), CODE,40,0,512, nullptr,0,0, 1,40,1,1, 1);
  conv(3, CODE,40,0,512,P(I_E0_W2),P(I_E0_B2), TC1,40,0,512,  nullptr,0,0, 40,40,1,1, 1);
  conv(5, TC1,40,0,512, P(I_SR_W),P(I_SR_B),   H1,40,0,256,   nullptr,0,0, 40,40,2,2, 1);
  conv(3, H1,40,0,256,  P(I_E1_W1),P(I_E1_B1), R50a,40,0,256, nullptr,0,0, 40,40,1,1, 1);
  conv(3, R50a,40,0,256,P(I_E1_W2),P(I_E1_B2), H2,40,0,256,   H1,40,0,     40,40,1,1, 1);
  conv(3, H2,40,0,256,  P(I_E2_W1),P(I_E2_B1), R50a,40,0,256, nullptr,0,0, 40,40,1,1, 1);
  conv(3, R50a,40,0,256,P(I_E2_W2),P(I_E2_B2), H1,40,0,256,   H2,40,0,     40,40,1,1, 1);
  // final h in H1 (B,40,256); cs = ch 0..19, cn = ch 20..39

  // ---- mid + code assign, branch s then branch n ----
  for (int br = 0; br < 2; ++br) {
    int coff = br ? 20 : 0;
    conv(3, H1,40,coff,256, P(I_M0_W1),P(I_M0_B1), A20,20,0,256, nullptr,0,0, 20,20,1,1, 1);
    conv(3, A20,20,0,256,   P(I_M0_W2),P(I_M0_B2), B20,20,0,256, H1,40,coff, 20,20,1,1, 1);
    conv(5, B20,20,0,256,   P(I_MC_W), P(I_MC_B),  A15,15,0,256, nullptr,0,0, 20,15,1,2, 1);
    conv(3, A15,15,0,256,   P(I_M1_W1),P(I_M1_B1), B15,15,0,256, nullptr,0,0, 15,15,1,1, 1);
    conv(3, B15,15,0,256,   P(I_M1_W2),P(I_M1_B2), C15,15,0,256, A15,15,0,   15,15,1,1, 1);
    if (br == 0) {
      fill0(C30s, (size_t)NB*30*256);
      code_assign_kernel<32><<<(NB*256+255)/256,256,0,stream>>>(
          C15, P(I_MEANS_S), PROB_S, C30s, NB, 256, 30, 15, 1.0f);  // [zeros, q]
    } else {
      fill0(C30n, (size_t)NB*30*256);
      code_assign_kernel<10><<<(NB*256+255)/256,256,0,stream>>>(
          C15, P(I_MEANS_N), PROB_N, C30n, NB, 256, 30, 0, 1.0f);   // [q, zeros]
    }
  }

  // ---- addin + sub_pixel into CODE (B,50,512) ----
  for (int br = 0; br < 2; ++br) {
    const float* cin30 = br ? C30n : C30s;
    conv(3, cin30,30,0,256, P(I_AI_W), P(I_AI_B),  R50a,50,0,256, nullptr,0,0, 30,50,1,1, 0);
    conv(3, R50a,50,0,256,  P(I_AB_W1),P(I_AB_B1), R50b,50,0,256, nullptr,0,0, 50,50,1,1, 1);
    conv(3, R50b,50,0,256,  P(I_AB_W2),P(I_AB_B2), R50c,50,0,256, R50a,50,0,  50,50,1,1, 1);
    long long tot = (long long)NB*25*256*2;
    sub_pixel_kernel<<<(int)((tot+255)/256),256,0,stream>>>(R50c, CODE, NB, 25, 256, 50, br ? 25 : 0);
  }

  // ---- ao0 / ao1 over (B,50,512) ----
  conv(3, CODE,50,0,512, P(I_AO0_W1),P(I_AO0_B1), TC1,50,0,512, nullptr,0,0, 50,50,1,1, 1);
  conv(3, TC1,50,0,512,  P(I_AO0_W2),P(I_AO0_B2), TC2,50,0,512, CODE,50,0,   50,50,1,1, 1);
  conv(3, TC2,50,0,512,  P(I_AO1_W1),P(I_AO1_B1), TC1,50,0,512, nullptr,0,0, 50,50,1,1, 1);
  conv(3, TC1,50,0,512,  P(I_AO1_W2),P(I_AO1_B2), CODE,50,0,512,TC2,50,0,    50,50,1,1, 1);
  // final code in CODE; cs = ch 0..24, cn = ch 25..49

  // ---- decoder per branch: d0, d1 blocks then FC+tanh ----
  for (int br = 0; br < 2; ++br) {
    int coff = br ? 25 : 0;
    conv(3, CODE,50,coff,512, P(I_D0_W1),P(I_D0_B1), R50a,25,0,512, nullptr,0,0, 25,25,1,1, 1);
    conv(3, R50a,25,0,512,    P(I_D0_W2),P(I_D0_B2), R50b,25,0,512, CODE,50,coff,25,25,1,1, 1);
    conv(3, R50b,25,0,512,    P(I_D1_W1),P(I_D1_B1), R50a,25,0,512, nullptr,0,0, 25,25,1,1, 1);
    conv(3, R50a,25,0,512,    P(I_D1_W2),P(I_D1_B2), R50c,25,0,512, R50b,25,0,  25,25,1,1, 1);
    // R50c viewed as (1024, 12800) row-major activations
    long long tiles = (long long)(NB/16) * (512/16);
    fc_tanh_wmma_bf16<<<(int)((tiles+7)/8),256,0,stream>>>(
        R50c, P(I_FC_W), P(I_FC_B), br ? DEC_N : DEC_S, NB, 12800, 512);
  }
  #undef P
}
